// BinarizeLinear_inference_19516331393234
// MI455X (gfx1250) — 8-bit matrix operand order repaired and checked by reading, compiled, not run on hardware
//
#include <hip/hip_runtime.h>
#include <hip/hip_bf16.h>


typedef __attribute__((ext_vector_type(8))) int v8i;

#define N_ROWS 16384
#define K_DIM  2048
#define J_DIM  2048

__global__ __launch_bounds__(256) void binarize_f32_to_i8(
    const float4* __restrict__ src, uint4* __restrict__ dst, int n16)
{
    int i = blockIdx.x * blockDim.x + threadIdx.x;
    if (i >= n16) return;
    unsigned int r[4];
#pragma unroll
    for (int j = 0; j < 4; ++j) {
        float4 f = src[i * 4 + j];
        unsigned int b0 = f.x > 0.0f ? 0x01u : 0xFFu;
        unsigned int b1 = f.y > 0.0f ? 0x01u : 0xFFu;
        unsigned int b2 = f.z > 0.0f ? 0x01u : 0xFFu;
        unsigned int b3 = f.w > 0.0f ? 0x01u : 0xFFu;
        r[j] = b0 | (b1 << 8) | (b2 << 16) | (b3 << 24);
    }
    uint4 o; o.x = r[0]; o.y = r[1]; o.z = r[2]; o.w = r[3];
    dst[i] = o;
}

__global__ __launch_bounds__(256) void bin_gemm_wmma_iu8(
    const signed char* __restrict__ xb,
    const signed char* __restrict__ wb,
    const float* __restrict__ bias,
    float* __restrict__ out)
{
    const int lane = threadIdx.x & 31;
    const int wid  = threadIdx.x >> 5;
    const int wm   = wid & 1;
    const int wn   = wid >> 1;

    const int n0 = blockIdx.y * 128 + wm * 64;
    const int j0 = blockIdx.x * 256 + wn * 64;

    const int half = lane >> 4;
    const int l16  = lane & 15;

    const signed char* aptr[4];
#pragma unroll
    for (int mi = 0; mi < 4; ++mi)
        aptr[mi] = xb + (size_t)(n0 + mi * 16 + l16) * K_DIM + half * 8;

    const signed char* bptr[4];
#pragma unroll
    for (int ni = 0; ni < 4; ++ni)
        bptr[ni] = wb + (size_t)(j0 + ni * 16 + l16) * K_DIM + half * 8;

    v8i acc[4][4] = {};

    for (int kk = 0; kk < K_DIM; kk += 64) {
        v8i a[4], b[4];
#pragma unroll
        for (int mi = 0; mi < 4; ++mi) {
#pragma unroll
            for (int g = 0; g < 4; ++g) {
                int2 d = *(const int2*)(aptr[mi] + kk + g * 16);
                a[mi][2 * g]     = d.x;
                a[mi][2 * g + 1] = d.y;
            }
        }
#pragma unroll
        for (int ni = 0; ni < 4; ++ni) {
#pragma unroll
            for (int g = 0; g < 4; ++g) {
                int2 d = *(const int2*)(bptr[ni] + kk + g * 16);
                b[ni][2 * g]     = d.x;
                b[ni][2 * g + 1] = d.y;
            }
        }

        if (kk + 64 < K_DIM) {
            __builtin_prefetch(aptr[0] + kk + 64, 0, 1);
            __builtin_prefetch(aptr[2] + kk + 64, 0, 1);
            __builtin_prefetch(bptr[0] + kk + 64, 0, 1);
            __builtin_prefetch(bptr[2] + kk + 64, 0, 1);
        }

#pragma unroll
        for (int mi = 0; mi < 4; ++mi)
#pragma unroll
            for (int ni = 0; ni < 4; ++ni)
                acc[mi][ni] = __builtin_amdgcn_wmma_i32_16x16x64_iu8(
                     true, a[mi],  true, b[ni],
                    acc[mi][ni],  false,  false);
    }

#pragma unroll
    for (int ni = 0; ni < 4; ++ni) {
        const int col = j0 + ni * 16 + l16;
        const float bv = bias[col];
#pragma unroll
        for (int mi = 0; mi < 4; ++mi) {
#pragma unroll
            for (int r = 0; r < 8; ++r) {
                const int row = n0 + mi * 16 + half * 8 + r;
                out[(size_t)row * J_DIM + col] = (float)acc[mi][ni][r] + bv;
            }
        }
    }
}

extern "C" void kernel_launch(void* const* d_in, const int* in_sizes, int n_in,
                              void* d_out, int out_size, void* d_ws, size_t ws_size,
                              hipStream_t stream)
{
    const float* x    = (const float*)d_in[0];
    const float* w    = (const float*)d_in[1];
    const float* bias = (const float*)d_in[2];
    float* out        = (float*)d_out;

    signed char* xb = (signed char*)d_ws;
    signed char* wb = xb + (size_t)N_ROWS * K_DIM;

    {
        int n16 = (N_ROWS * K_DIM) / 16;
        binarize_f32_to_i8<<<(n16 + 255) / 256, 256, 0, stream>>>(
            (const float4*)x, (uint4*)xb, n16);
    }
    {
        int n16 = (J_DIM * K_DIM) / 16;
        binarize_f32_to_i8<<<(n16 + 255) / 256, 256, 0, stream>>>(
            (const float4*)w, (uint4*)wb, n16);
    }

    dim3 grid(J_DIM / 256, N_ROWS / 128);
    bin_gemm_wmma_iu8<<<grid, 256, 0, stream>>>(xb, wb, bias, out);
}
